// CNN_Casual_21552145891462
// MI455X (gfx1250) — matrix operand order repaired and checked by reading, not compiled, not run on hardware
//
#include <hip/hip_runtime.h>
#include <math.h>

typedef __attribute__((ext_vector_type(16))) _Float16 v16h;
typedef __attribute__((ext_vector_type(8)))  _Float16 v8h;
typedef __attribute__((ext_vector_type(8)))  float    v8f;

#define WMMA_F16(a, b, c) \
  __builtin_amdgcn_wmma_f32_16x16x32_f16(false, (a), false, (b), (short)0, (c), false, false)

// ---- workspace layout (bytes) ----
static constexpr size_t SIG_OFF = 0;                    // 784 f32  sigmoid(mask)
static constexpr size_t A1_OFF  = 4096;                 // 512 f16  conv1 A frag (bias in k=25)
static constexpr size_t A2_OFF  = A1_OFF + 1024;        // 8192 f16 conv2 A frags [2 mt][8 ks] (bias in k=250)
static constexpr size_t BF1_OFF = A2_OFF + 16384;       // 20480 f16 fc1 B frags [4 nt][10 ks]
static constexpr size_t BF2_OFF = BF1_OFF + 40960;      // 1024 f16 fc2 B frags [2 ks]
static constexpr size_t H1_OFF  = 65536;                // B*1440 f16, then B*320 f16 (h2)

// A-fragment slot->k (16-bit A 16x32): lanes0-15 K=0..15, lanes16-31 K=16..31.
__device__ __host__ __forceinline__ int akmap(int h, int j) {
  return h * 16 + j;
}
// B-fragment slot->k (16-bit B 32x16): lanes0-15 slots0-7=K0-7, slots8-15=K16-23;
// lanes16-31 slots0-7=K8-15, slots8-15=K24-31.
__device__ __host__ __forceinline__ int bkmap(int h, int j) { return (j < 8) ? (h * 8 + j) : (16 + h * 8 + (j - 8)); }

// two aligned b128 LDS loads -> one A fragment (slots 0..7 and 8..15 are runs of 8 k)
__device__ __forceinline__ v16h load_afrag(const _Float16* base_lo) {
  v8h lo = *(const v8h*)(base_lo);
  v8h hi = *(const v8h*)(base_lo + 16);
  return __builtin_shufflevector(lo, hi, 0, 1, 2, 3, 4, 5, 6, 7,
                                 8, 9, 10, 11, 12, 13, 14, 15);
}

// ---------------- weight / mask packing ----------------
__global__ void prep_kernel(const float* mask_w, const float* c1w, const float* c1b,
                            const float* c2w, const float* c2b,
                            const float* f1w, const float* f2w, char* ws) {
  const int tid = threadIdx.x;  // single block of 256
  float*     sig = (float*)(ws + SIG_OFF);
  _Float16*  A1  = (_Float16*)(ws + A1_OFF);
  _Float16*  A2  = (_Float16*)(ws + A2_OFF);
  _Float16*  B1  = (_Float16*)(ws + BF1_OFF);
  _Float16*  B2  = (_Float16*)(ws + BF2_OFF);

  for (int i = tid; i < 784; i += 256) sig[i] = 1.f / (1.f + __expf(-mask_w[i]));

  // conv1 A: M=16(10 valid) x K=32; k<25 weights, k==25 bias, else 0
  for (int t = tid; t < 512; t += 256) {
    int l = t >> 4, j = t & 15;
    int m = l & 15, k = akmap(l >> 4, j);
    float v = 0.f;
    if (m < 10) v = (k < 25) ? c1w[m * 25 + k] : ((k == 25) ? c1b[m] : 0.f);
    A1[t] = (_Float16)v;
  }
  // conv2 A: [mt 0..1][ks 0..7]; k<250 weights, k==250 bias, else 0
  for (int t = tid; t < 16 * 512; t += 256) {
    int f = t >> 9, l = (t >> 4) & 31, j = t & 15;
    int mt = f >> 3, ks = f & 7;
    int ch = mt * 16 + (l & 15);
    int k  = ks * 32 + akmap(l >> 4, j);
    float v = 0.f;
    if (ch < 20) v = (k < 250) ? c2w[ch * 250 + k] : ((k == 250) ? c2b[ch] : 0.f);
    A2[t] = (_Float16)v;
  }
  // fc1 B: [nt 0..3][ks 0..9]; B[k][n] = fc1_w[n][k], K=320 exact, n>=50 zero
  for (int t = tid; t < 40 * 512; t += 256) {
    int f = t >> 9, l = (t >> 4) & 31, j = t & 15;
    int nt = f / 10, ks = f % 10;
    int n = nt * 16 + (l & 15);
    int k = ks * 32 + bkmap(l >> 4, j);
    B1[t] = (_Float16)((n < 50) ? f1w[n * 320 + k] : 0.f);
  }
  // fc2 B: [ks 0..1]; B[k][n] = fc2_w[n][k], K pad 50->64, N pad 10->16
  for (int t = tid; t < 2 * 512; t += 256) {
    int f = t >> 9, l = (t >> 4) & 31, j = t & 15;
    int n = l & 15;
    int k = f * 32 + bkmap(l >> 4, j);
    B2[t] = (_Float16)((n < 10 && k < 50) ? f2w[n * 50 + k] : 0.f);
  }
}

// ---------------- conv1 (+bias via GEMM) + maxpool + relu ----------------
__global__ __launch_bounds__(128) void conv1_kernel(const float* __restrict__ x,
                                                    char* ws, _Float16* __restrict__ h1) {
  const int lane = threadIdx.x & 31;
  const int w    = threadIdx.x >> 5;
  const int img  = blockIdx.x * 4 + w;
  const float*    sig = (const float*)(ws + SIG_OFF);
  const _Float16* A1  = (const _Float16*)(ws + A1_OFF);

  __shared__ _Float16 sx[4][800];    // masked input (784) + sentinel 1.0 at [784]
  __shared__ _Float16 sc[4][6336];   // conv out [10][576] + 1 dummy row

  {
    const float4* xv = (const float4*)(x + (size_t)img * 784);
    const float4* sv = (const float4*)sig;
    for (int i = lane; i < 196; i += 32) {   // 196 float4 = 784
      float4 a = xv[i], s = sv[i];
      union { _Float16 e[4]; unsigned long long u; } o;
      o.e[0] = (_Float16)(a.x * s.x);
      o.e[1] = (_Float16)(a.y * s.y);
      o.e[2] = (_Float16)(a.z * s.z);
      o.e[3] = (_Float16)(a.w * s.w);
      *(unsigned long long*)&sx[w][i * 4] = o.u;
    }
  }
  if (lane == 0) sx[w][784] = (_Float16)1.f;   // bias sentinel
  if (blockIdx.x + 1 < gridDim.x)
    __builtin_prefetch(x + (size_t)(img + 4) * 784, 0, 3);
  __syncthreads();

  const v16h a  = *(const v16h*)(A1 + lane * 16);
  const int  h  = lane >> 4;
  const int  pl = lane & 15;
  const v8f  zero = {0.f, 0.f, 0.f, 0.f, 0.f, 0.f, 0.f, 0.f};

  // per-slot im2col offsets (depend only on half-wave), clamped for pad slots
  int off[16];
#pragma unroll
  for (int j = 0; j < 16; ++j) {
    int k  = h * 16 + j;
    int kk = (k < 25) ? k : 0;
    off[j] = (kk / 5) * 28 + (kk % 5);
  }

  int p  = pl;    // output pixel index = nt*16 + pl
  int ow = pl;    // column within 24 (for incremental pb update)
  int pb = pl;    // input base = oh*28 + ow
  for (int nt = 0; nt < 36; ++nt) {
    union { _Float16 e[16]; v16h v; } b;
#pragma unroll
    for (int j = 0; j < 16; ++j) {
      int idx = (h * 16 + j == 25) ? 784 : (pb + off[j]);  // cndmask on index
      b.e[j] = sx[w][idx];
    }
    v8f acc = zero;
    acc = WMMA_F16(a, b.v, acc);
#pragma unroll
    for (int r = 0; r < 8; ++r) {
      int chs = min(r + 8 * h, 10);      // invalid rows -> dummy row 10
      sc[w][chs * 576 + p] = (_Float16)acc[r];
    }
    // incremental p/pb update (no divisions)
    p += 16;
    ow += 16;
    int wrap = (ow >= 24);
    pb += wrap ? 20 : 16;
    ow -= wrap ? 24 : 0;
  }
  __syncthreads();

  // 2x2 maxpool + relu, 2 outputs per lane (b64 LDS loads, b32 global store)
  for (int pi = lane; pi < 720; pi += 32) {
    int i = pi * 2;
    int ch = i / 144, rem = i % 144;
    int pr = rem / 12, pc = rem % 12;    // pc even
    const _Float16* cp = &sc[w][ch * 576 + pr * 48 + pc * 2];
    union { _Float16 e[4]; unsigned long long u; } r0, r1;
    r0.u = *(const unsigned long long*)cp;
    r1.u = *(const unsigned long long*)(cp + 24);
    union { _Float16 e[2]; unsigned u; } o;
    o.e[0] = (_Float16)fmaxf(fmaxf(fmaxf((float)r0.e[0], (float)r0.e[1]),
                                   fmaxf((float)r1.e[0], (float)r1.e[1])), 0.f);
    o.e[1] = (_Float16)fmaxf(fmaxf(fmaxf((float)r0.e[2], (float)r0.e[3]),
                                   fmaxf((float)r1.e[2], (float)r1.e[3])), 0.f);
    *(unsigned*)(h1 + (size_t)img * 1440 + i) = o.u;
  }
}

// ---------------- conv2 (+bias via GEMM) + maxpool + relu ----------------
__global__ __launch_bounds__(128) void conv2_kernel(char* ws,
                                                    const _Float16* __restrict__ h1,
                                                    _Float16* __restrict__ h2) {
  const int lane = threadIdx.x & 31;
  const int w    = threadIdx.x >> 5;
  const int img  = blockIdx.x * 4 + w;
  const _Float16* A2 = (const _Float16*)(ws + A2_OFF);

  __shared__ _Float16 s1[4][1456];   // h1 image [10][12][12] + sentinel 1.0 at [1440]
  __shared__ _Float16 s2[4][1344];   // conv out [20][64] + 1 dummy row
  {
    const uint4* src = (const uint4*)(h1 + (size_t)img * 1440);
    uint4* dst = (uint4*)&s1[w][0];
    for (int i = lane; i < 180; i += 32) dst[i] = src[i];   // 180 uint4 = 1440 f16
  }
  if (lane == 0) s1[w][1440] = (_Float16)1.f;   // bias sentinel
  if (blockIdx.x + 1 < gridDim.x)
    __builtin_prefetch(h1 + (size_t)(img + 4) * 1440, 0, 3);
  __syncthreads();

  const int h  = lane >> 4;
  const int pl = lane & 15;
  const v8f zero = {0.f, 0.f, 0.f, 0.f, 0.f, 0.f, 0.f, 0.f};
  v8f acc[4][2];
#pragma unroll
  for (int nt = 0; nt < 4; ++nt) { acc[nt][0] = zero; acc[nt][1] = zero; }

  const int pb0 = (pl >> 3) * 12 + (pl & 7);   // base for pixel pl; +24 per n-tile

  for (int ks = 0; ks < 8; ++ks) {
    v16h a0 = *(const v16h*)(A2 + ((size_t)((0 * 8 + ks) * 32 + lane)) * 16);
    v16h a1 = *(const v16h*)(A2 + ((size_t)((1 * 8 + ks) * 32 + lane)) * 16);
    const int kbase = ks * 32 + h * 16;
    int off[16];
#pragma unroll
    for (int j = 0; j < 16; ++j) {
      int k  = kbase + j;
      int kk = (k < 250) ? k : 0;
      int ic = kk / 25, rem = kk % 25;
      off[j] = ic * 144 + (rem / 5) * 12 + (rem % 5);
    }
#pragma unroll
    for (int nt = 0; nt < 4; ++nt) {
      int pb = pb0 + 24 * nt;
      union { _Float16 e[16]; v16h v; } b;
#pragma unroll
      for (int j = 0; j < 16; ++j) {
        int idx = (kbase + j == 250) ? 1440 : (pb + off[j]);  // cndmask on index
        b.e[j] = s1[w][idx];
      }
      acc[nt][0] = WMMA_F16(a0, b.v, acc[nt][0]);
      acc[nt][1] = WMMA_F16(a1, b.v, acc[nt][1]);
    }
  }
#pragma unroll
  for (int nt = 0; nt < 4; ++nt) {
    int p = nt * 16 + pl;
#pragma unroll
    for (int mt = 0; mt < 2; ++mt) {
#pragma unroll
      for (int r = 0; r < 8; ++r) {
        int chs = min(mt * 16 + r + 8 * h, 20);  // invalid rows -> dummy row 20
        s2[w][chs * 64 + p] = (_Float16)acc[nt][mt][r];
      }
    }
  }
  __syncthreads();

  // 2x2 maxpool + relu -> h2[img][320], 2 outputs per lane
  for (int pi = lane; pi < 160; pi += 32) {
    int i = pi * 2;
    int ch = i >> 4, pos = i & 15;
    int pr = pos >> 2, pc = pos & 3;     // pc in {0,2}
    const _Float16* cp = &s2[w][ch * 64 + pr * 16 + pc * 2];
    union { _Float16 e[4]; unsigned long long u; } r0, r1;
    r0.u = *(const unsigned long long*)cp;
    r1.u = *(const unsigned long long*)(cp + 8);
    union { _Float16 e[2]; unsigned u; } o;
    o.e[0] = (_Float16)fmaxf(fmaxf(fmaxf((float)r0.e[0], (float)r0.e[1]),
                                   fmaxf((float)r1.e[0], (float)r1.e[1])), 0.f);
    o.e[1] = (_Float16)fmaxf(fmaxf(fmaxf((float)r0.e[2], (float)r0.e[3]),
                                   fmaxf((float)r1.e[2], (float)r1.e[3])), 0.f);
    *(unsigned*)(h2 + (size_t)img * 320 + i) = o.u;
  }
}

// ---------------- fc1+relu, fc2, log_softmax ----------------
__global__ __launch_bounds__(128) void fc_kernel(const float* __restrict__ f1b,
                                                 const float* __restrict__ f2b,
                                                 float* __restrict__ out, char* ws,
                                                 const _Float16* __restrict__ h2) {
  const int lane = threadIdx.x & 31;
  const int w    = threadIdx.x >> 5;
  const int tile = blockIdx.x * 4 + w;    // images tile*16 .. tile*16+15
  const _Float16* B1 = (const _Float16*)(ws + BF1_OFF);
  const _Float16* B2 = (const _Float16*)(ws + BF2_OFF);

  __shared__ _Float16 sA[4][5120];  // 16 x 320 activations
  __shared__ _Float16 sF[4][1024];  // 16 x 64 fc1 out (cols 50..63 are zero)
  __shared__ float    sL[4][256];   // 16 x 16 logits
  {
    const uint4* src = (const uint4*)(h2 + (size_t)tile * 5120);
    uint4* dst = (uint4*)&sA[w][0];
    for (int i = lane; i < 640; i += 32) dst[i] = src[i];   // 640 uint4 = 5120 f16
  }
  if (blockIdx.x + 1 < gridDim.x)
    __builtin_prefetch(h2 + (size_t)(tile + 4) * 5120, 0, 3);

  const int h  = lane >> 4;
  const int nl = lane & 15;
  // hoisted biases (exec-masked loads happen before any WMMA)
  float fb1[4];
#pragma unroll
  for (int nt = 0; nt < 4; ++nt) {
    int n = nt * 16 + nl;
    fb1[nt] = (n < 50) ? f1b[n] : 0.f;
  }
  const float fb2 = (nl < 10) ? f2b[nl] : 0.f;
  __syncthreads();

  const v8f zero = {0.f, 0.f, 0.f, 0.f, 0.f, 0.f, 0.f, 0.f};
  v8f acc[4];
#pragma unroll
  for (int nt = 0; nt < 4; ++nt) acc[nt] = zero;

  // fc1: (16x320) x (320x64), 10 ksteps; A fragment = two ds_load_b128
  for (int ks = 0; ks < 10; ++ks) {
    v16h av = load_afrag(&sA[w][nl * 320 + ks * 32 + h * 8]);
#pragma unroll
    for (int nt = 0; nt < 4; ++nt) {
      v16h bf = *(const v16h*)(B1 + ((size_t)((nt * 10 + ks) * 32 + lane)) * 16);
      acc[nt] = WMMA_F16(av, bf, acc[nt]);
    }
  }
  // bias + relu -> sF  (n>=50: acc==0 and bias==0 -> stores 0, keeps K-pad clean)
#pragma unroll
  for (int nt = 0; nt < 4; ++nt) {
    int n = nt * 16 + nl;
#pragma unroll
    for (int r = 0; r < 8; ++r) {
      int m = r + 8 * h;
      sF[w][m * 64 + n] = (_Float16)fmaxf(acc[nt][r] + fb1[nt], 0.f);
    }
  }
  __syncthreads();

  // fc2: (16x64) x (64x16), 2 ksteps; A fragment = two ds_load_b128
  v8f accL = zero;
  for (int ks = 0; ks < 2; ++ks) {
    v16h av = load_afrag(&sF[w][nl * 64 + ks * 32 + h * 8]);
    v16h bf = *(const v16h*)(B2 + ((size_t)(ks * 32 + lane)) * 16);
    accL = WMMA_F16(av, bf, accL);
  }
#pragma unroll
  for (int r = 0; r < 8; ++r) {
    int m = r + 8 * h;
    sL[w][m * 16 + nl] = accL[r] + fb2;
  }
  __syncthreads();

  // log_softmax per image row (lanes 0..15 each own one image)
  if (lane < 16) {
    const float* row = &sL[w][lane * 16];
    float mx = row[0];
#pragma unroll
    for (int j = 1; j < 10; ++j) mx = fmaxf(mx, row[j]);
    float s = 0.f;
#pragma unroll
    for (int j = 0; j < 10; ++j) s += __expf(row[j] - mx);
    float lse = mx + __logf(s);
    float* o = out + ((size_t)tile * 16 + lane) * 10;
#pragma unroll
    for (int j = 0; j < 10; ++j) o[j] = row[j] - lse;
  }
}

extern "C" void kernel_launch(void* const* d_in, const int* in_sizes, int n_in,
                              void* d_out, int out_size, void* d_ws, size_t ws_size,
                              hipStream_t stream) {
  const float* x      = (const float*)d_in[0];
  const float* mask_w = (const float*)d_in[1];
  const float* c1w    = (const float*)d_in[2];
  const float* c1b    = (const float*)d_in[3];
  const float* c2w    = (const float*)d_in[4];
  const float* c2b    = (const float*)d_in[5];
  const float* f1w    = (const float*)d_in[6];
  const float* f1b    = (const float*)d_in[7];
  const float* f2w    = (const float*)d_in[8];
  const float* f2b    = (const float*)d_in[9];
  float* out = (float*)d_out;
  char*  ws  = (char*)d_ws;

  const int B = in_sizes[0] / 784;  // 8192
  _Float16* h1 = (_Float16*)(ws + H1_OFF);
  _Float16* h2 = (_Float16*)(ws + H1_OFF + (size_t)B * 1440 * 2);

  prep_kernel<<<1, 256, 0, stream>>>(mask_w, c1w, c1b, c2w, c2b, f1w, f2w, ws);
  conv1_kernel<<<B / 4, 128, 0, stream>>>(x, ws, h1);
  conv2_kernel<<<B / 4, 128, 0, stream>>>(ws, h1, h2);
  fc_kernel<<<B / 64, 128, 0, stream>>>(f1b, f2b, out, ws, h2);
}
